// MuMURMultimodalAttention_8315056685637
// MI455X (gfx1250) — matrix operand order repaired and checked by reading, not compiled, not run on hardware
//
#include <hip/hip_runtime.h>
#include <hip/hip_bf16.h>
#include <math.h>

// ---------------------------------------------------------------------------
// Problem constants
// ---------------------------------------------------------------------------
#define DMODEL   1024
#define NHEAD    16
#define HDIM     64
#define NTEXT    256
#define NVISION  256
#define NFRAMES  12
#define VCHUNK   32          // vision samples per chunk (8 chunks)

typedef __attribute__((ext_vector_type(16))) __bf16 v16bf;
typedef __attribute__((ext_vector_type(8)))  __bf16 v8bf;
typedef __attribute__((ext_vector_type(8)))  float  v8f;
typedef __attribute__((ext_vector_type(4)))  float  f4;
typedef __attribute__((ext_vector_type(4)))  unsigned u4;

static __device__ __forceinline__ unsigned fbits(float x) {
    return __builtin_bit_cast(unsigned, x);
}
// pack two fp32 (lo,hi) -> dword of two bf16 via one v_perm_b32
// sel 0x07060302: out[1:0]=src1.byte[3:2] (lo), out[3:2]=src0.byte[3:2] (hi)
static __device__ __forceinline__ unsigned pkbf(float lo, float hi) {
    return __builtin_amdgcn_perm(fbits(hi), fbits(lo), 0x07060302u);
}

// ---------------------------------------------------------------------------
// fp32 -> packed bf16 conversion (truncation), 8 floats / thread.
// nPairs must be a multiple of 4*blockDim (true for all sizes here).
// ---------------------------------------------------------------------------
__global__ __launch_bounds__(256) void cvt_f32_bf16(
    const float* __restrict__ src, unsigned* __restrict__ dst /*bf16 pairs*/)
{
    const size_t i = (size_t)blockIdx.x * blockDim.x + threadIdx.x;
    const f4 x0 = ((const f4*)src)[2 * i + 0];
    const f4 x1 = ((const f4*)src)[2 * i + 1];
    u4 p;
    p.x = pkbf(x0[0], x0[1]);
    p.y = pkbf(x0[2], x0[3]);
    p.z = pkbf(x1[0], x1[1]);
    p.w = pkbf(x1[2], x1[3]);
    ((u4*)dst)[i] = p;
}

// ---------------------------------------------------------------------------
// GEMM: C[M,N] = A[M,K]bf16 * (W[N,K]bf16)^T + bias[N]f32, C in f32.
// One wave computes a 32(M) x 64(N) tile: 2 A operands x 4 B operands ->
// 8x v_wmma_f32_16x16x32_bf16 per 32-wide k-step. No conversion in the loop.
//
// Register layouts used here (wave32; A and B share one k order):
//   A (16x32 bf16): lane L -> row m=L&15; element i holds
//                   K = 16*(L>>4) + i  => two 16B chunks/lane.
//   B (32x16 bf16): lane L -> column n=L&15 (W row n0+n); element i holds
//                   K = 16*(L>>4) + i  => one contiguous 32B chunk (2x16B).
//   C/D (16x16 f32): lane L, vgpr r -> (m = r + 8*(L>>4), n = L&15).
// Requires: M%32==0, N%64==0, K%32==0, totalTiles%8==0 (256-thread blocks).
// ---------------------------------------------------------------------------
__global__ __launch_bounds__(256) void gemm_bf16_wmma(
    const __bf16* __restrict__ A,
    const __bf16* __restrict__ W,
    const float*  __restrict__ bias,
    float*        __restrict__ C,
    int M, int N, int K)
{
    const int lane = threadIdx.x & 31;
    const int wave = blockIdx.x * 8 + (threadIdx.x >> 5);

    const int tilesN = N >> 6;
    const int tileM  = wave / tilesN;
    const int tileN  = wave - tileM * tilesN;
    if (tileM * 32 >= M) return;               // wave-uniform; EXEC stays full

    const int half = lane >> 4;
    const int mr   = lane & 15;
    const int m0   = tileM * 32;
    const int n0   = tileN * 64;

    const __bf16* Ar0 = A + (size_t)(m0 +  0 + mr) * K;
    const __bf16* Ar1 = A + (size_t)(m0 + 16 + mr) * K;
    const __bf16* Wr0 = W + (size_t)(n0 +  0 + mr) * K;
    const __bf16* Wr1 = W + (size_t)(n0 + 16 + mr) * K;
    const __bf16* Wr2 = W + (size_t)(n0 + 32 + mr) * K;
    const __bf16* Wr3 = W + (size_t)(n0 + 48 + mr) * K;

    v8f acc0 = {}, acc1 = {}, acc2 = {}, acc3 = {};
    v8f acc4 = {}, acc5 = {}, acc6 = {}, acc7 = {};

#define SHUF16(lo, hi) __builtin_shufflevector(lo, hi, 0,1,2,3,4,5,6,7,8,9,10,11,12,13,14,15)

    for (int k0 = 0; k0 < K; k0 += 32) {
        // A operands: two 8-bf16 (16B) chunks per lane
        const v8bf a0l = *(const v8bf*)(Ar0 + k0 +     16 * half);
        const v8bf a0h = *(const v8bf*)(Ar0 + k0 + 8 + 16 * half);
        const v8bf a1l = *(const v8bf*)(Ar1 + k0 +     16 * half);
        const v8bf a1h = *(const v8bf*)(Ar1 + k0 + 8 + 16 * half);
        const v16bf av0 = SHUF16(a0l, a0h);
        const v16bf av1 = SHUF16(a1l, a1h);
        // B operands: 16 contiguous bf16 (32B) per lane, 4 N-subtiles
        const int kb = k0 + 16 * half;
        const v16bf bv0 = SHUF16(*(const v8bf*)(Wr0 + kb), *(const v8bf*)(Wr0 + kb + 8));
        const v16bf bv1 = SHUF16(*(const v8bf*)(Wr1 + kb), *(const v8bf*)(Wr1 + kb + 8));
        const v16bf bv2 = SHUF16(*(const v8bf*)(Wr2 + kb), *(const v8bf*)(Wr2 + kb + 8));
        const v16bf bv3 = SHUF16(*(const v8bf*)(Wr3 + kb), *(const v8bf*)(Wr3 + kb + 8));

        acc0 = __builtin_amdgcn_wmma_f32_16x16x32_bf16(false, av0, false, bv0, (short)0, acc0, false, false);
        acc1 = __builtin_amdgcn_wmma_f32_16x16x32_bf16(false, av0, false, bv1, (short)0, acc1, false, false);
        acc2 = __builtin_amdgcn_wmma_f32_16x16x32_bf16(false, av0, false, bv2, (short)0, acc2, false, false);
        acc3 = __builtin_amdgcn_wmma_f32_16x16x32_bf16(false, av0, false, bv3, (short)0, acc3, false, false);
        acc4 = __builtin_amdgcn_wmma_f32_16x16x32_bf16(false, av1, false, bv0, (short)0, acc4, false, false);
        acc5 = __builtin_amdgcn_wmma_f32_16x16x32_bf16(false, av1, false, bv1, (short)0, acc5, false, false);
        acc6 = __builtin_amdgcn_wmma_f32_16x16x32_bf16(false, av1, false, bv2, (short)0, acc6, false, false);
        acc7 = __builtin_amdgcn_wmma_f32_16x16x32_bf16(false, av1, false, bv3, (short)0, acc7, false, false);
    }
#undef SHUF16

    const float b0 = bias[n0 +  0 + mr];
    const float b1 = bias[n0 + 16 + mr];
    const float b2 = bias[n0 + 32 + mr];
    const float b3 = bias[n0 + 48 + mr];
    #pragma unroll
    for (int r = 0; r < 8; ++r) {
        const int mA = m0 + r + 8 * half;          // M-subtile 0
        const int mB = mA + 16;                    // M-subtile 1
        float* ca = C + (size_t)mA * N + n0 + mr;
        float* cb = C + (size_t)mB * N + n0 + mr;
        ca[0]  = acc0[r] + b0;  ca[16] = acc1[r] + b1;
        ca[32] = acc2[r] + b2;  ca[48] = acc3[r] + b3;
        cb[0]  = acc4[r] + b0;  cb[16] = acc5[r] + b1;
        cb[32] = acc6[r] + b2;  cb[48] = acc7[r] + b3;
    }
}

// ---------------------------------------------------------------------------
// Attention core: per (v, h) block, 256 threads (one per text token t).
//   scores[f,t] = <K[v,f,h,:], Q[t,h,:]> / 8 ; softmax over frames (12)
//   attn_bf16[vLocal, t, h*64+d] = sum_f probs[f,t] * V[v,f,h,d]
// K/V tiles (12x64 f32) staged in LDS; output packed to bf16 via v_perm.
// ---------------------------------------------------------------------------
__global__ __launch_bounds__(256) void attn_softmax_ctx(
    const float* __restrict__ Q,      // [NTEXT, DMODEL] f32
    const float* __restrict__ Kp,     // [NVISION*NFRAMES, DMODEL] f32
    const float* __restrict__ Vp,     // [NVISION*NFRAMES, DMODEL] f32
    unsigned*    __restrict__ attn,   // [VCHUNK*NTEXT, DMODEL] bf16 (as pairs)
    int vBase)
{
    const int vLocal = blockIdx.x;
    const int v      = vBase + vLocal;
    const int h      = blockIdx.y;
    const int t      = threadIdx.x;

    __shared__ float Ks[NFRAMES][HDIM];
    __shared__ float Vs[NFRAMES][HDIM];

    for (int i = threadIdx.x; i < NFRAMES * HDIM; i += blockDim.x) {
        const int f = i >> 6, d = i & 63;
        const size_t row = (size_t)(v * NFRAMES + f) * DMODEL + h * HDIM;
        Ks[f][d] = Kp[row + d];
        Vs[f][d] = Vp[row + d];
    }
    __syncthreads();

    float s[NFRAMES];
    #pragma unroll
    for (int f = 0; f < NFRAMES; ++f) s[f] = 0.f;
    const float* qrow = Q + (size_t)t * DMODEL + h * HDIM;
    for (int d = 0; d < HDIM; ++d) {
        const float qd = qrow[d];
        #pragma unroll
        for (int f = 0; f < NFRAMES; ++f) s[f] = fmaf(Ks[f][d], qd, s[f]);
    }

    float mx = -INFINITY;
    #pragma unroll
    for (int f = 0; f < NFRAMES; ++f) { s[f] *= 0.125f; mx = fmaxf(mx, s[f]); }
    float sum = 0.f;
    #pragma unroll
    for (int f = 0; f < NFRAMES; ++f) { s[f] = __expf(s[f] - mx); sum += s[f]; }
    const float inv = 1.f / sum;

    // context -> bf16, 128 contiguous bytes per thread (8x b128 stores)
    unsigned* orow = attn + (((size_t)vLocal * NTEXT + t) * DMODEL + h * HDIM) / 2;
    #pragma unroll
    for (int d = 0; d < HDIM; d += 8) {
        float o[8];
        #pragma unroll
        for (int j = 0; j < 8; ++j) o[j] = 0.f;
        #pragma unroll
        for (int f = 0; f < NFRAMES; ++f) {
            const float p = s[f];
            #pragma unroll
            for (int j = 0; j < 8; ++j) o[j] = fmaf(p, Vs[f][d + j], o[j]);
        }
        u4 pk;
        pk.x = pkbf(o[0] * inv, o[1] * inv);
        pk.y = pkbf(o[2] * inv, o[3] * inv);
        pk.z = pkbf(o[4] * inv, o[5] * inv);
        pk.w = pkbf(o[6] * inv, o[7] * inv);
        *(u4*)(orow + d / 2) = pk;
    }
}

// ---------------------------------------------------------------------------
// Launcher
// ---------------------------------------------------------------------------
extern "C" void kernel_launch(void* const* d_in, const int* /*in_sizes*/, int /*n_in*/,
                              void* d_out, int /*out_size*/, void* d_ws, size_t /*ws_size*/,
                              hipStream_t stream)
{
    const float* text   = (const float*)d_in[0];   // [256,1024]
    const float* vision = (const float*)d_in[1];   // [256,12,1024] -> [3072,1024]
    const float* wq = (const float*)d_in[2];
    const float* bq = (const float*)d_in[3];
    const float* wk = (const float*)d_in[4];
    const float* bk = (const float*)d_in[5];
    const float* wv = (const float*)d_in[6];
    const float* bv = (const float*)d_in[7];
    const float* wo = (const float*)d_in[8];
    const float* bo = (const float*)d_in[9];
    float* out = (float*)d_out;

    const size_t DD   = (size_t)DMODEL * DMODEL;           // 1M elements
    const size_t NVF  = (size_t)NVISION * NFRAMES;         // 3072 rows

    // ---- workspace layout ----
    char* w = (char*)d_ws;
    float*  qf32  = (float*)w;                 w += (size_t)NTEXT * DMODEL * 4;   //  1.0 MB
    float*  kf32  = (float*)w;                 w += NVF * DMODEL * 4;             // 12.6 MB
    float*  vf32  = (float*)w;                 w += NVF * DMODEL * 4;             // 12.6 MB
    __bf16* textb = (__bf16*)w;                w += (size_t)NTEXT * DMODEL * 2;   //  0.5 MB
    __bf16* visb  = (__bf16*)w;                w += NVF * DMODEL * 2;             //  6.3 MB
    __bf16* wqb   = (__bf16*)w;                w += DD * 2;                       //  2.0 MB
    __bf16* wkb   = (__bf16*)w;                w += DD * 2;
    __bf16* wvb   = (__bf16*)w;                w += DD * 2;
    __bf16* wob   = (__bf16*)w;                w += DD * 2;
    __bf16* attnb = (__bf16*)w;                /* 8192*1024 bf16 = 16.8 MB */

    // ---- fp32 -> bf16 conversions (8 floats / thread, 256-thread blocks) ----
    #define CVT(src, dst, nElem) \
        cvt_f32_bf16<<<(unsigned)((nElem) / 8 / 256), 256, 0, stream>>>(src, (unsigned*)(dst))
    CVT(text,   textb, (size_t)NTEXT * DMODEL);
    CVT(vision, visb,  NVF * DMODEL);
    CVT(wq, wqb, DD);
    CVT(wk, wkb, DD);
    CVT(wv, wvb, DD);
    CVT(wo, wob, DD);
    #undef CVT

    // ---- projections (bf16 x bf16 -> f32) ----
    // tiles = (M/32) * (N/64); 8 waves (256 thr) per block
    gemm_bf16_wmma<<< (NTEXT/32) * (DMODEL/64) / 8, 256, 0, stream>>>(
        textb, wqb, bq, qf32, NTEXT, DMODEL, DMODEL);
    gemm_bf16_wmma<<< (int)(NVF/32) * (DMODEL/64) / 8, 256, 0, stream>>>(
        visb, wkb, bk, kf32, (int)NVF, DMODEL, DMODEL);
    gemm_bf16_wmma<<< (int)(NVF/32) * (DMODEL/64) / 8, 256, 0, stream>>>(
        visb, wvb, bv, vf32, (int)NVF, DMODEL, DMODEL);

    // ---- attention + output projection, chunked over vision batch ----
    const int Mc = VCHUNK * NTEXT;                          // 8192 rows
    for (int c = 0; c < NVISION / VCHUNK; ++c) {
        const int vBase = c * VCHUNK;
        dim3 agrid(VCHUNK, NHEAD);
        attn_softmax_ctx<<<agrid, 256, 0, stream>>>(qf32, kf32, vf32,
                                                    (unsigned*)attnb, vBase);
        float* outChunk = out + (size_t)vBase * NTEXT * DMODEL;
        gemm_bf16_wmma<<< (Mc/32) * (DMODEL/64) / 8, 256, 0, stream>>>(
            attnb, wob, bo, outChunk, Mc, DMODEL, DMODEL);
    }
}
